// MambaAttention_80212809220547
// MI455X (gfx1250) — compile-verified, matrix operand order checked by reading, not run on hardware
//
#include <hip/hip_runtime.h>
#include <hip/hip_bf16.h>

// ---------------- problem constants (from reference) ----------------
#define BB     2
#define LL     2048
#define DM     512          // D_MODEL
#define DI     1024         // D_INNER
#define DS     16           // D_STATE
#define DCONV  4
#define DTR    32           // DT_RANK
#define ML     (BB * LL)    // 4096 rows for all GEMMs

// GEMM blocking
#define BKK    32           // K-step (one bf16 WMMA)
#define BM_    128          // 8 waves * 16 rows
#define BN_    64           // 4 x 16-wide WMMA tiles per wave

// ---------------- types ----------------
typedef __attribute__((ext_vector_type(16))) __bf16          bf16x16;
typedef __attribute__((ext_vector_type(8)))  float           f32x8;
typedef __attribute__((ext_vector_type(8)))  unsigned short  us8;
typedef __attribute__((ext_vector_type(4)))  unsigned int    u32x4;
typedef __attribute__((ext_vector_type(8)))  int             i32x8;
typedef __attribute__((ext_vector_type(4)))  int             i32x4;

union FragU { bf16x16 v; us8 h[2]; };

#if defined(__AMDGCN__) && defined(__has_builtin)
#  if __has_builtin(__builtin_amdgcn_tensor_load_to_lds)
#    define HAVE_TDM 1
#  endif
#endif
#ifndef HAVE_TDM
#  define HAVE_TDM 0
#endif

__device__ __forceinline__ unsigned short f32_to_bf16(float f) {
  unsigned int u = __builtin_bit_cast(unsigned int, f);
  u += 0x7FFFu + ((u >> 16) & 1u);            // round-to-nearest-even
  return (unsigned short)(u >> 16);
}

__device__ __forceinline__ float sigmoidf_(float x) {
  return 1.0f / (1.0f + __expf(-x));
}

// ---------------- TDM: async 2D bf16 tile (rows x 32) global -> LDS ----------------
// Descriptor per CDNA5 ISA:
//   group0: count=1 | lds_addr | global_addr[56:0] | type=2
//   group1: data_size=1(2B) | tensor_dim0/1 | tile_dim0=32, tile_dim1=rows |
//           tensor_dim0_stride = pitch (elements)
__device__ __forceinline__ void tdm_load_tile(const unsigned short* gsrc,
                                              unsigned lds_off,
                                              int rows, int pitch_elems) {
#if HAVE_TDM
  unsigned long long ga = (unsigned long long)(const void*)gsrc;
  u32x4 g0;
  g0.x = 1u;                                                  // count=1, user desc
  g0.y = lds_off;                                             // lds_addr (bytes)
  g0.z = (unsigned)ga;                                        // global_addr[31:0]
  g0.w = (unsigned)((ga >> 32) & 0x01FFFFFFu) | (2u << 30);   // addr[56:32] | type=2
  const unsigned td0 = (unsigned)pitch_elems;                 // tensor_dim0
  const unsigned td1 = (unsigned)rows;                        // tensor_dim1
  const unsigned long long st0 = (unsigned long long)pitch_elems; // dim0 stride
  i32x8 g1;
  g1[0] = (int)(1u << 16);                                    // data_size=1 (2 bytes)
  g1[1] = (int)((td0 & 0xFFFFu) << 16);                       // tensor_dim0[15:0]
  g1[2] = (int)((td0 >> 16) | ((td1 & 0xFFFFu) << 16));       // dim0 hi | dim1 lo
  g1[3] = (int)((td1 >> 16) | ((unsigned)BKK << 16));         // dim1 hi | tile_dim0
  g1[4] = (int)(td1 & 0xFFFFu);                               // tile_dim1 | tile_dim2=0
  g1[5] = (int)(unsigned)(st0 & 0xFFFFFFFFu);                 // dim0_stride[31:0]
  g1[6] = (int)(unsigned)((st0 >> 32) & 0xFFFFu);             // dim0_stride[47:32]
  g1[7] = 0;                                                  // dim1_stride (2D: unused)
  const i32x4 gz = (i32x4){0, 0, 0, 0};
#if __clang_major__ >= 23
  const i32x8 gz8 = (i32x8){0, 0, 0, 0, 0, 0, 0, 0};
  __builtin_amdgcn_tensor_load_to_lds(g0, g1, gz, gz, gz8, 0);
#else
  __builtin_amdgcn_tensor_load_to_lds(g0, g1, gz, gz, 0);
#endif
#else
  (void)gsrc; (void)lds_off; (void)rows; (void)pitch_elems;
#endif
}

// Fallback: cooperative global->LDS copy of a (ROWS x 32) bf16 tile.
template <int ROWS>
__device__ __forceinline__ void coop_copy_tile(unsigned short* dst,
                                               const unsigned short* src,
                                               int ld, int tid, int nthreads) {
  const int chunks = ROWS * (BKK / 8);
  for (int i = tid; i < chunks; i += nthreads) {
    const int row = i >> 2;
    const int c   = i & 3;
    ((us8*)dst)[i] = *(const us8*)(src + (long long)row * ld + c * 8);
  }
}

// ---------------- WMMA GEMM:  C[M,N] = A[M,K] * W[N,K]^T ----------------
// Block = 256 threads (8 waves) -> 128x64 C tile. A/B staged in LDS through the
// Tensor Data Mover (double-buffered, TENSORcnt-pipelined). Wave w owns M-tile w.
// EPI: 0 = plain store, 1 = bias + softplus (dt projection).
template <int EPI>
__global__ void wmma_gemm_kernel(const unsigned short* __restrict__ A,
                                 const unsigned short* __restrict__ W,
                                 float* __restrict__ C,
                                 const float* __restrict__ bias,
                                 int M, int N, int K) {
  __shared__ unsigned short sA[2][BM_ * BKK];   // 2 x 8 KB
  __shared__ unsigned short sB[2][BN_ * BKK];   // 2 x 4 KB

  const int tid  = (int)threadIdx.x;
  const int lane = tid & 31;
  const int wid  = tid >> 5;                    // wave in block: 0..7
  const int half = lane >> 4;
  const int r    = lane & 15;

  const int nblk = N / BN_;
  const int bm   = (int)blockIdx.x / nblk;
  const int bn   = (int)blockIdx.x % nblk;

  const unsigned short* Abase = A + (long long)(bm * BM_) * K;
  const unsigned short* Wbase = W + (long long)(bn * BN_) * K;

  f32x8 acc[4];
#pragma unroll
  for (int t = 0; t < 4; ++t) acc[t] = (f32x8){0.f,0.f,0.f,0.f,0.f,0.f,0.f,0.f};

  const int nsteps = K / BKK;

#if HAVE_TDM
  // prologue: stage 0 into buffer 0 (wave 0 drives the DMA for the block)
  if (wid == 0) {
    tdm_load_tile(Abase, (unsigned)(unsigned long long)(void*)&sA[0][0], BM_, K);
    tdm_load_tile(Wbase, (unsigned)(unsigned long long)(void*)&sB[0][0], BN_, K);
  }
#else
  coop_copy_tile<BM_>(&sA[0][0], Abase, K, tid, (int)blockDim.x);
  coop_copy_tile<BN_>(&sB[0][0], Wbase, K, tid, (int)blockDim.x);
#endif

  for (int s = 0; s < nsteps; ++s) {
    const int cur = s & 1;
    // stage s+1 into the other buffer while we compute on `cur`
    if (s + 1 < nsteps) {
#if HAVE_TDM
      if (wid == 0) {
        tdm_load_tile(Abase + (s + 1) * BKK,
                      (unsigned)(unsigned long long)(void*)&sA[cur ^ 1][0], BM_, K);
        tdm_load_tile(Wbase + (s + 1) * BKK,
                      (unsigned)(unsigned long long)(void*)&sB[cur ^ 1][0], BN_, K);
      }
#else
      coop_copy_tile<BM_>(&sA[cur ^ 1][0], Abase + (s + 1) * BKK, K, tid, (int)blockDim.x);
      coop_copy_tile<BN_>(&sB[cur ^ 1][0], Wbase + (s + 1) * BKK, K, tid, (int)blockDim.x);
#endif
    }
#if HAVE_TDM
    // TDM ops from one wave complete in order: <=2 outstanding means stage s landed.
    if (s + 1 < nsteps) __builtin_amdgcn_s_wait_tensorcnt(2);
    else                __builtin_amdgcn_s_wait_tensorcnt(0);
#endif
    __syncthreads();

    // A fragment: lanes 0-15 hold K 0-7 / 16-23 of row r, lanes 16-31 K 8-15 / 24-31
    FragU fa;
    {
      const unsigned short* pa = &sA[cur][(wid * 16 + r) * BKK + half * 8];
      fa.h[0] = *(const us8*)pa;
      fa.h[1] = *(const us8*)(pa + 16);
    }
#pragma unroll
    for (int t = 0; t < 4; ++t) {
      FragU fb;
      const unsigned short* pb = &sB[cur][(t * 16 + r) * BKK + half * 8];
      fb.h[0] = *(const us8*)pb;
      fb.h[1] = *(const us8*)(pb + 16);
      acc[t] = __builtin_amdgcn_wmma_f32_16x16x32_bf16(
          /*neg_a=*/false, fa.v, /*neg_b=*/false, fb.v,
          /*c_mod=*/(short)0, acc[t], /*reuse_a=*/false, /*reuse_b=*/false);
    }
    __syncthreads();   // everyone done with `cur` before it is re-staged
  }

  // C/D layout: VGPR j -> M = half*8 + j, N = lane&15
#pragma unroll
  for (int t = 0; t < 4; ++t) {
    const int ncol = bn * BN_ + t * 16 + r;
#pragma unroll
    for (int j = 0; j < 8; ++j) {
      const int mrow = bm * BM_ + wid * 16 + half * 8 + j;
      float v = acc[t][j];
      if (EPI == 1) {
        v += bias[ncol];
        v = (v > 20.f) ? v : log1pf(__expf(v));   // softplus
      }
      C[(long long)mrow * N + ncol] = v;
    }
  }
}

// ---------------- f32 -> bf16 convert (with optional column slice) ----------------
__global__ void cvt_bf16_kernel(const float* __restrict__ src,
                                unsigned short* __restrict__ dst,
                                int rows, int src_cols, int col0, int ncols) {
  long long idx = (long long)blockIdx.x * blockDim.x + threadIdx.x;
  long long total = (long long)rows * ncols;
  if (idx >= total) return;
  int r = (int)(idx / ncols);
  int c = (int)(idx % ncols);
  dst[idx] = f32_to_bf16(src[(long long)r * src_cols + col0 + c]);
}

// ---------------- depthwise causal conv (K=4) + bias + SiLU ----------------
__global__ void conv_silu_kernel(const float* __restrict__ xz,
                                 const float* __restrict__ conv_w,
                                 const float* __restrict__ conv_b,
                                 float* __restrict__ xi_conv,
                                 unsigned short* __restrict__ xi_bf) {
  long long idx = (long long)blockIdx.x * blockDim.x + threadIdx.x;
  if (idx >= (long long)ML * DI) return;
  const int d = (int)(idx & (DI - 1));
  const int m = (int)(idx >> 10);
  const int l = m & (LL - 1);
  const int b = m >> 11;
  float acc = conv_b[d];
#pragma unroll
  for (int k = 0; k < DCONV; ++k) {
    const int ls = l - (DCONV - 1) + k;
    if (ls >= 0)
      acc += conv_w[d * DCONV + k] * xz[(long long)(b * LL + ls) * (2 * DI) + d];
  }
  const float s = acc * sigmoidf_(acc);   // SiLU
  xi_conv[idx] = s;
  xi_bf[idx]   = f32_to_bf16(s);
}

// ---------------- selective scan ----------------
// one thread per (b, d, n); 16 lanes (one n-group) share (b,d).
__global__ void scan_kernel(const float* __restrict__ dt,
                            const float* __restrict__ xi,
                            const float* __restrict__ xdbl,   // [ML, 64]: dt_r|Bm|Cm
                            const float* __restrict__ A_log,
                            float* __restrict__ yout) {
  const int t  = (int)(blockIdx.x * blockDim.x + threadIdx.x);
  if (t >= BB * DI * DS) return;
  const int n  = t & (DS - 1);
  const int bd = t >> 4;
  const int d  = bd & (DI - 1);
  const int b  = bd >> 10;

  const float Av = -__expf(A_log[d * DS + n]);
  float h = 0.f;

  for (int l = 0; l < LL; ++l) {
    const int m = b * LL + l;
    if (l + 1 < LL)   // pull next step's B/C line early -> global_prefetch_b8
      __builtin_prefetch(xdbl + (long long)(m + 1) * 64 + DTR, 0, 1);
    const float dtv = dt[(long long)m * DI + d];
    const float xiv = xi[(long long)m * DI + d];
    const float Bv  = xdbl[(long long)m * 64 + DTR + n];
    const float Cv  = xdbl[(long long)m * 64 + DTR + DS + n];
    const float dA  = __expf(dtv * Av);
    h = dA * h + (dtv * xiv) * Bv;
    float p = h * Cv;
    p += __shfl_xor(p, 1, 16);
    p += __shfl_xor(p, 2, 16);
    p += __shfl_xor(p, 4, 16);
    p += __shfl_xor(p, 8, 16);
    if (n == 0) yout[(long long)m * DI + d] = p;
  }
}

// ---------------- gate: y = (ys + D*xi) * silu(z), emit bf16 ----------------
__global__ void gate_kernel(const float* __restrict__ yscan,
                            const float* __restrict__ xi_conv,
                            const float* __restrict__ D_param,
                            const float* __restrict__ xz,
                            unsigned short* __restrict__ ybf) {
  long long idx = (long long)blockIdx.x * blockDim.x + threadIdx.x;
  if (idx >= (long long)ML * DI) return;
  const int d = (int)(idx & (DI - 1));
  const int m = (int)(idx >> 10);
  const float yv = yscan[idx] + D_param[d] * xi_conv[idx];
  const float zv = xz[(long long)m * (2 * DI) + DI + d];
  ybf[idx] = f32_to_bf16(yv * (zv * sigmoidf_(zv)));
}

// =====================================================================
extern "C" void kernel_launch(void* const* d_in, const int* in_sizes, int n_in,
                              void* d_out, int out_size, void* d_ws, size_t ws_size,
                              hipStream_t stream) {
  (void)in_sizes; (void)n_in; (void)out_size; (void)ws_size;

  const float* q       = (const float*)d_in[0];   // queries (B,L,H,E) == (ML, DM)
  const float* W_in    = (const float*)d_in[3];   // (2*DI, DM)
  const float* conv_w  = (const float*)d_in[4];   // (DI, 1, DCONV)
  const float* conv_b  = (const float*)d_in[5];   // (DI,)
  const float* W_xproj = (const float*)d_in[6];   // (64, DI)
  const float* W_dt    = (const float*)d_in[7];   // (DI, DTR)
  const float* b_dt    = (const float*)d_in[8];   // (DI,)
  const float* A_log   = (const float*)d_in[9];   // (DI, DS)
  const float* D_param = (const float*)d_in[10];  // (DI,)
  const float* W_out   = (const float*)d_in[11];  // (DM, DI)
  float* out = (float*)d_out;                     // (ML, DM)

  // ---- workspace layout ----
  char* ws = (char*)d_ws;
  size_t o = 0;
  auto alloc = [&](size_t bytes) { size_t r = o; o += (bytes + 255) & ~(size_t)255; return r; };
  const size_t o_xbf   = alloc((size_t)ML * DM * 2);
  const size_t o_winb  = alloc((size_t)2 * DI * DM * 2);
  const size_t o_xz    = alloc((size_t)ML * 2 * DI * 4);
  const size_t o_xic   = alloc((size_t)ML * DI * 4);
  const size_t o_xibf  = alloc((size_t)ML * DI * 2);
  const size_t o_wxpb  = alloc((size_t)(DTR + 2 * DS) * DI * 2);
  const size_t o_xdbl  = alloc((size_t)ML * (DTR + 2 * DS) * 4);
  const size_t o_dtrb  = alloc((size_t)ML * DTR * 2);
  const size_t o_wdtb  = alloc((size_t)DI * DTR * 2);
  const size_t o_dt    = alloc((size_t)ML * DI * 4);
  const size_t o_ysc   = alloc((size_t)ML * DI * 4);
  const size_t o_ybf   = alloc((size_t)ML * DI * 2);
  const size_t o_woutb = alloc((size_t)DM * DI * 2);

  unsigned short* xbf   = (unsigned short*)(ws + o_xbf);
  unsigned short* winb  = (unsigned short*)(ws + o_winb);
  float*          xz    = (float*)(ws + o_xz);
  float*          xic   = (float*)(ws + o_xic);
  unsigned short* xibf  = (unsigned short*)(ws + o_xibf);
  unsigned short* wxpb  = (unsigned short*)(ws + o_wxpb);
  float*          xdbl  = (float*)(ws + o_xdbl);
  unsigned short* dtrb  = (unsigned short*)(ws + o_dtrb);
  unsigned short* wdtb  = (unsigned short*)(ws + o_wdtb);
  float*          dtb   = (float*)(ws + o_dt);
  float*          ysc   = (float*)(ws + o_ysc);
  unsigned short* ybf   = (unsigned short*)(ws + o_ybf);
  unsigned short* woutb = (unsigned short*)(ws + o_woutb);

  const int TB = 256;
  auto blocks = [&](long long n) { return (unsigned)((n + TB - 1) / TB); };
  auto gemm_grid = [&](int M, int N) { return (unsigned)((M / BM_) * (N / BN_)); };

  // 1) converts (weights + activations to bf16)
  cvt_bf16_kernel<<<blocks((long long)ML * DM), TB, 0, stream>>>(q, xbf, ML, DM, 0, DM);
  cvt_bf16_kernel<<<blocks((long long)2 * DI * DM), TB, 0, stream>>>(W_in, winb, 2 * DI, DM, 0, DM);
  cvt_bf16_kernel<<<blocks((long long)64 * DI), TB, 0, stream>>>(W_xproj, wxpb, DTR + 2 * DS, DI, 0, DI);
  cvt_bf16_kernel<<<blocks((long long)DI * DTR), TB, 0, stream>>>(W_dt, wdtb, DI, DTR, 0, DTR);
  cvt_bf16_kernel<<<blocks((long long)DM * DI), TB, 0, stream>>>(W_out, woutb, DM, DI, 0, DI);

  // 2) in-projection: xz[4096,2048] = x * W_in^T
  wmma_gemm_kernel<0><<<gemm_grid(ML, 2 * DI), TB, 0, stream>>>(
      xbf, winb, xz, nullptr, ML, 2 * DI, DM);

  // 3) depthwise causal conv + SiLU
  conv_silu_kernel<<<blocks((long long)ML * DI), TB, 0, stream>>>(xz, conv_w, conv_b, xic, xibf);

  // 4) x-projection: x_dbl[4096,64] = xi * W_xproj^T
  wmma_gemm_kernel<0><<<gemm_grid(ML, DTR + 2 * DS), TB, 0, stream>>>(
      xibf, wxpb, xdbl, nullptr, ML, DTR + 2 * DS, DI);

  // 5) slice dt_r columns to bf16
  cvt_bf16_kernel<<<blocks((long long)ML * DTR), TB, 0, stream>>>(xdbl, dtrb, ML, DTR + 2 * DS, 0, DTR);

  // 6) dt projection + bias + softplus
  wmma_gemm_kernel<1><<<gemm_grid(ML, DI), TB, 0, stream>>>(
      dtrb, wdtb, dtb, b_dt, ML, DI, DTR);

  // 7) selective scan (serial in L, parallel over b*d*n = 32768 threads)
  scan_kernel<<<blocks((long long)BB * DI * DS), TB, 0, stream>>>(dtb, xic, xdbl, A_log, ysc);

  // 8) gate: y = (ys + D*xi) * silu(z) -> bf16
  gate_kernel<<<blocks((long long)ML * DI), TB, 0, stream>>>(ysc, xic, D_param, xz, ybf);

  // 9) out-projection: out[4096,512] = y * W_out^T
  wmma_gemm_kernel<0><<<gemm_grid(ML, DM), TB, 0, stream>>>(
      ybf, woutb, out, nullptr, ML, DM, DI);
}
